// DecGAT_14568529068218
// MI455X (gfx1250) — compile-verified, matrix operand order checked by reading, not run on hardware
//
#include <hip/hip_runtime.h>
#include <hip/hip_bf16.h>

// ---------------------------------------------------------------------------
// DecGAT on MI455X (gfx1250, wave32).
//  - Dense GEMMs: v_wmma_f32_16x16x32_bf16. BT (N x K, bf16) staged once per
//    workgroup in LDS (4 waves/block, cooperative b128 copy + barrier);
//    K-loop: A-frag = 2x global_load_b128, B-frag = 2x ds_load_b128 (rows
//    padded +8 halves to rotate banks). One wave = 16x(16*NT) C strip.
//  - Edge phase: wave32 per edge, float4 (b128) gathers (L2-resident),
//    shfl_xor reductions, native global_atomic_add_f32 scatters.
// ---------------------------------------------------------------------------

typedef __attribute__((ext_vector_type(16))) __bf16 v16bf;
typedef __attribute__((ext_vector_type(8)))  __bf16 v8bf;
typedef __attribute__((ext_vector_type(8)))  float  v8f;

static __device__ __forceinline__ __bf16 f2bf(float f) {
  // round-to-nearest-even f32 -> bf16
  unsigned u = __builtin_bit_cast(unsigned, f);
  unsigned r = (u + 0x7FFFu + ((u >> 16) & 1u)) >> 16;
  return __builtin_bit_cast(__bf16, (unsigned short)r);
}

// ---------------------------------------------------------------------------
// C[M,N] = Abf[M,K] @ (BTbf[N,K])^T + bias[N]
// Abf has Mtiles*16 zero-padded rows. K mult of 32, N = 16*NT (gridDim.x==1).
// blockDim.x == 128 (4 waves, one 16-row M-tile each).
// Dynamic LDS: N * (K+8) * 2 bytes (BT rows padded 8 halves vs bank conflicts).
// Per-lane frag layout (16-bit, 16x32): halves [0..7] = K 8*grp..+7,
// halves [8..15] = K 16+8*grp..+7  (grp = lane>>4).
// Cbf (optional) gets a bf16 copy of C with pad rows zeroed.
// ---------------------------------------------------------------------------
template <int NT>
__global__ void gemm_bf16_wmma(const __bf16* __restrict__ Abf,
                               const __bf16* __restrict__ BTbf,
                               const float* __restrict__ bias,
                               float* __restrict__ C,
                               __bf16* __restrict__ Cbf,
                               int M, int K, int N, int Mtiles) {
  extern __shared__ __bf16 Blds[];          // N rows x (K+8) halves
  const int tid  = threadIdx.x;
  const int wave = tid >> 5;
  const int lane = tid & 31;
  const int grp  = lane >> 4;
  const int l16  = lane & 15;
  const int ldk  = K + 8;

  // cooperative stage of BT into LDS: 128 threads x 8 halves per step
  for (int base = tid * 8; base < N * K; base += 128 * 8) {
    const int row = base / K;
    const int col = base - row * K;         // multiple of 8
    *(v8bf*)(Blds + row * ldk + col) = *(const v8bf*)(BTbf + (size_t)row * K + col);
  }
  __syncthreads();

  const int tm = blockIdx.y * 4 + wave;
  if (tm >= Mtiles) return;                 // no more barriers below

  const __bf16* Arow = Abf + (size_t)(tm * 16 + l16) * K;   // padded: no guard

  v8f c[NT];
#pragma unroll
  for (int nt = 0; nt < NT; ++nt) c[nt] = (v8f){};

  for (int k0 = 0; k0 < K; k0 += 32) {
    if (k0 + 32 < K)
      __builtin_prefetch(Arow + k0 + 32, 0, 3);    // global_prefetch_b8

    const int kA = k0 + 8 * grp;
    const v8bf alo = *(const v8bf*)(Arow + kA);
    const v8bf ahi = *(const v8bf*)(Arow + kA + 16);
    v16bf a;
#pragma unroll
    for (int i = 0; i < 8; ++i) { a[i] = alo[i]; a[8 + i] = ahi[i]; }

#pragma unroll
    for (int nt = 0; nt < NT; ++nt) {
      const __bf16* Bl = Blds + (nt * 16 + l16) * ldk + kA;   // ds_load_b128
      const v8bf blo = *(const v8bf*)(Bl);
      const v8bf bhi = *(const v8bf*)(Bl + 16);
      v16bf b;
#pragma unroll
      for (int i = 0; i < 8; ++i) { b[i] = blo[i]; b[8 + i] = bhi[i]; }
      c[nt] = __builtin_amdgcn_wmma_f32_16x16x32_bf16(
          false, a, false, b, (short)0, c[nt], false, false);
    }
  }

#pragma unroll
  for (int nt = 0; nt < NT; ++nt) {
    const int col = nt * 16 + l16;
    const float bv = bias[col];
#pragma unroll
    for (int r = 0; r < 8; ++r) {
      const int row = tm * 16 + r + 8 * grp;   // C layout: VGPR r -> M=r (+8 hi half)
      const float val = c[nt][r] + bv;
      if (row < M) C[(size_t)row * N + col] = val;
      if (Cbf) Cbf[(size_t)row * N + col] = (row < M) ? f2bf(val) : f2bf(0.f);
    }
  }
}

// ---------------------- conversion / transpose helpers ---------------------
// dst[0..n) = bf16(src); dst[n..ntot) = 0   (4 elements / thread)
__global__ void f32_to_bf16_pad(const float* __restrict__ src,
                                __bf16* __restrict__ dst, long n, long ntot) {
  const long i = 4L * (blockIdx.x * (long)blockDim.x + threadIdx.x);
  if (i >= ntot) return;
  if (i + 3 < n) {
    const float4 v = *(const float4*)(src + i);
    dst[i]     = f2bf(v.x);
    dst[i + 1] = f2bf(v.y);
    dst[i + 2] = f2bf(v.z);
    dst[i + 3] = f2bf(v.w);
  } else {
#pragma unroll
    for (int j = 0; j < 4; ++j) {
      const long p = i + j;
      if (p < ntot) dst[p] = (p < n) ? f2bf(src[p]) : f2bf(0.f);
    }
  }
}

// BT[n,k] = bf16(B[k,n]);  B is K x N (small weight matrices)
__global__ void transpose_to_bf16(const float* __restrict__ B,
                                  __bf16* __restrict__ BT, int K, int N) {
  const int idx = blockIdx.x * blockDim.x + threadIdx.x;
  if (idx >= K * N) return;
  const int k = idx / N, n = idx - k * N;
  BT[(size_t)n * K + k] = f2bf(B[idx]);
}

// --------------------------- wave32 reduction ------------------------------
template <int NV>
static __device__ __forceinline__ void wave_reduce(float* v) {
#pragma unroll
  for (int off = 16; off > 0; off >>= 1) {
#pragma unroll
    for (int i = 0; i < NV; ++i) v[i] += __shfl_xor(v[i], off, 32);
  }
}

// na[n,0:2] = softmax(z[n,:] @ W[128,2]); one wave per node.
__global__ void node_softmax2(const float* __restrict__ z,
                              const float* __restrict__ W,
                              float* __restrict__ na, int n) {
  const int wid  = (blockIdx.x * blockDim.x + threadIdx.x) >> 5;
  const int lane = threadIdx.x & 31;
  if (wid >= n) return;
  const float4 zv = *(const float4*)(z + (size_t)wid * 128 + lane * 4);
  const float4 w0 = *(const float4*)(W + 8 * lane);
  const float4 w1 = *(const float4*)(W + 8 * lane + 4);
  float v[2];
  v[0] = zv.x * w0.x + zv.y * w0.z + zv.z * w1.x + zv.w * w1.z;
  v[1] = zv.x * w0.y + zv.y * w0.w + zv.z * w1.y + zv.w * w1.w;
  wave_reduce<2>(v);
  const float mx = fmaxf(v[0], v[1]);
  const float e0 = __expf(v[0] - mx), e1 = __expf(v[1] - mx);
  const float inv = 1.f / (e0 + e1);
  if (lane == 0) {
    na[(size_t)wid * 2 + 0] = e0 * inv;
    na[(size_t)wid * 2 + 1] = e1 * inv;
  }
}

// ee[e,0:2] = softmax(concat(z[src], z[dst]) @ W[256,2]); one wave per edge.
__global__ void edge_softmax2(const float* __restrict__ z,
                              const float* __restrict__ W,
                              const int* __restrict__ es,
                              const int* __restrict__ ed,
                              float* __restrict__ ee, int E) {
  const int wid  = (blockIdx.x * blockDim.x + threadIdx.x) >> 5;
  const int lane = threadIdx.x & 31;
  if (wid >= E) return;
  const int s = es[wid], d = ed[wid];
  const float4 zs = *(const float4*)(z + (size_t)s * 128 + lane * 4);
  const float4 zd = *(const float4*)(z + (size_t)d * 128 + lane * 4);
  const float4 w0 = *(const float4*)(W + 8 * lane);
  const float4 w1 = *(const float4*)(W + 8 * lane + 4);
  const float4 w2 = *(const float4*)(W + 256 + 8 * lane);
  const float4 w3 = *(const float4*)(W + 256 + 8 * lane + 4);
  float v[2];
  v[0] = zs.x * w0.x + zs.y * w0.z + zs.z * w1.x + zs.w * w1.z +
         zd.x * w2.x + zd.y * w2.z + zd.z * w3.x + zd.w * w3.z;
  v[1] = zs.x * w0.y + zs.y * w0.w + zs.z * w1.y + zs.w * w1.w +
         zd.x * w2.y + zd.y * w2.w + zd.z * w3.y + zd.w * w3.w;
  wave_reduce<2>(v);
  const float mx = fmaxf(v[0], v[1]);
  const float e0 = __expf(v[0] - mx), e1 = __expf(v[1] - mx);
  const float inv = 1.f / (e0 + e1);
  if (lane == 0) {
    ee[(size_t)wid * 2 + 0] = e0 * inv;
    ee[(size_t)wid * 2 + 1] = e1 * inv;
  }
}

// ei[e] = sigmoid(concat(z[src], z[dst]) @ W[256,1]); one wave per edge.
__global__ void edge_sigmoid(const float* __restrict__ z,
                             const float* __restrict__ W,
                             const int* __restrict__ es,
                             const int* __restrict__ ed,
                             float* __restrict__ ei, int E) {
  const int wid  = (blockIdx.x * blockDim.x + threadIdx.x) >> 5;
  const int lane = threadIdx.x & 31;
  if (wid >= E) return;
  const int s = es[wid], d = ed[wid];
  const float4 zs = *(const float4*)(z + (size_t)s * 128 + lane * 4);
  const float4 zd = *(const float4*)(z + (size_t)d * 128 + lane * 4);
  const float4 ws = *(const float4*)(W + 4 * lane);
  const float4 wd = *(const float4*)(W + 128 + 4 * lane);
  float v[1];
  v[0] = zs.x * ws.x + zs.y * ws.y + zs.z * ws.z + zs.w * ws.w +
         zd.x * wd.x + zd.y * wd.y + zd.z * wd.z + zd.w * wd.w;
  wave_reduce<1>(v);
  if (lane == 0) ei[wid] = 1.f / (1.f + __expf(-v[0]));
}

// Fused e-channel scatter: s[dst] += ee0*wz[src]; eh[dst] += ee1*z[src]; deg[dst]+=1.
__global__ void scatter_e(const float* __restrict__ z,
                          const float* __restrict__ wz,
                          const float* __restrict__ ee,
                          const int* __restrict__ es,
                          const int* __restrict__ ed,
                          float* __restrict__ sacc,
                          float* __restrict__ eh,
                          float* __restrict__ deg, int E) {
  const int wid  = (blockIdx.x * blockDim.x + threadIdx.x) >> 5;
  const int lane = threadIdx.x & 31;
  if (wid >= E) return;
  const int s = es[wid], d = ed[wid];
  const float w0 = ee[(size_t)wid * 2 + 0];
  const float w1 = ee[(size_t)wid * 2 + 1];
  const float4 wzv = *(const float4*)(wz + (size_t)s * 128 + lane * 4);
  const float4 zv  = *(const float4*)(z  + (size_t)s * 128 + lane * 4);
  float* sp = sacc + (size_t)d * 128 + lane * 4;
  float* ep = eh   + (size_t)d * 128 + lane * 4;
  unsafeAtomicAdd(sp + 0, w0 * wzv.x);
  unsafeAtomicAdd(sp + 1, w0 * wzv.y);
  unsafeAtomicAdd(sp + 2, w0 * wzv.z);
  unsafeAtomicAdd(sp + 3, w0 * wzv.w);
  unsafeAtomicAdd(ep + 0, w1 * zv.x);
  unsafeAtomicAdd(ep + 1, w1 * zv.y);
  unsafeAtomicAdd(ep + 2, w1 * zv.z);
  unsafeAtomicAdd(ep + 3, w1 * zv.w);
  if (lane == 0) unsafeAtomicAdd(&deg[d], 1.0f);
}

// i-channel scatter: ih[dst] += ei*z[src].
__global__ void scatter_i(const float* __restrict__ z,
                          const float* __restrict__ ei,
                          const int* __restrict__ is_,
                          const int* __restrict__ id_,
                          float* __restrict__ ih, int E) {
  const int wid  = (blockIdx.x * blockDim.x + threadIdx.x) >> 5;
  const int lane = threadIdx.x & 31;
  if (wid >= E) return;
  const int s = is_[wid], d = id_[wid];
  const float w = ei[wid];
  const float4 zv = *(const float4*)(z + (size_t)s * 128 + lane * 4);
  float* ip = ih + (size_t)d * 128 + lane * 4;
  unsafeAtomicAdd(ip + 0, w * zv.x);
  unsafeAtomicAdd(ip + 1, w * zv.y);
  unsafeAtomicAdd(ip + 2, w * zv.z);
  unsafeAtomicAdd(ip + 3, w * zv.w);
}

// z_out = 0.5*( relu(s/max(deg,1)) + na0*z + relu(eh*ih) + na1*z ); 4 elem/thread
__global__ void finalize_layer(const float* __restrict__ z,
                               const float* __restrict__ na,
                               const float* __restrict__ sacc,
                               const float* __restrict__ deg,
                               const float* __restrict__ eh,
                               const float* __restrict__ ih,
                               float* __restrict__ zout, int n_dst) {
  const int t = blockIdx.x * blockDim.x + threadIdx.x;   // over n_dst*32
  if (t >= n_dst * 32) return;
  const int idx = t * 4;
  const int n = idx >> 7;
  const float4 zv = *(const float4*)(z + idx);
  const float4 sv = *(const float4*)(sacc + idx);
  const float4 ev = *(const float4*)(eh + idx);
  const float4 iv = *(const float4*)(ih + idx);
  const float invd = 1.f / fmaxf(deg[n], 1.0f);
  const float a0 = na[(size_t)n * 2 + 0], a1 = na[(size_t)n * 2 + 1];
  float4 o;
  o.x = 0.5f * (fmaxf(sv.x * invd, 0.f) + a0 * zv.x + fmaxf(ev.x * iv.x, 0.f) + a1 * zv.x);
  o.y = 0.5f * (fmaxf(sv.y * invd, 0.f) + a0 * zv.y + fmaxf(ev.y * iv.y, 0.f) + a1 * zv.y);
  o.z = 0.5f * (fmaxf(sv.z * invd, 0.f) + a0 * zv.z + fmaxf(ev.z * iv.z, 0.f) + a1 * zv.z);
  o.w = 0.5f * (fmaxf(sv.w * invd, 0.f) + a0 * zv.w + fmaxf(ev.w * iv.w, 0.f) + a1 * zv.w);
  *(float4*)(zout + idx) = o;
}

__global__ void zero_f32(float* __restrict__ p, int n) {   // n mult of 4
  const int i = 4 * (blockIdx.x * blockDim.x + threadIdx.x);
  if (i < n) *(float4*)(p + i) = (float4){0.f, 0.f, 0.f, 0.f};
}

// ---------------------------------------------------------------------------
extern "C" void kernel_launch(void* const* d_in, const int* in_sizes, int n_in,
                              void* d_out, int out_size, void* d_ws, size_t ws_size,
                              hipStream_t stream) {
  const int F = 256, D = 128, O = 64;
  const int NS0 = in_sizes[0] / F;            // 100000 (mult of 16)
  const int E0  = in_sizes[1];                // 500000
  const int E1  = in_sizes[5];                // 100000
  const int ND0 = 25000, ND1 = 5000;          // reference scalars
  const int NS1 = ND0;
  auto ceil16 = [](int m) { return (m + 15) & ~15; };
  const int NS1p = ceil16(NS1);               // 25008
  const int ND1p = ceil16(ND1);               // 5008

  const float* feat = (const float*)d_in[0];
  const int* src0_cor = (const int*)d_in[1];
  const int* dst0_cor = (const int*)d_in[2];
  const int* src0_sim = (const int*)d_in[3];
  const int* dst0_sim = (const int*)d_in[4];
  const int* src1_cor = (const int*)d_in[5];
  const int* dst1_cor = (const int*)d_in[6];
  const int* src1_sim = (const int*)d_in[7];
  const int* dst1_sim = (const int*)d_in[8];
  const float* Win  = (const float*)d_in[11];
  const float* b_in = (const float*)d_in[12];
  const float* na1  = (const float*)d_in[13];
  const float* ea1  = (const float*)d_in[14];
  const float* ia1  = (const float*)d_in[15];
  const float* W1   = (const float*)d_in[16];
  const float* b1   = (const float*)d_in[17];
  const float* na2  = (const float*)d_in[18];
  const float* ea2  = (const float*)d_in[19];
  const float* ia2  = (const float*)d_in[20];
  const float* W2   = (const float*)d_in[21];
  const float* b2   = (const float*)d_in[22];
  const float* Wout = (const float*)d_in[23];
  const float* bout = (const float*)d_in[24];
  float* out = (float*)d_out;

  // ---- workspace carve-out (256B aligned) ----
  char* ws = (char*)d_ws;
  size_t off = 0;
  auto carve = [&](size_t bytes) -> void* {
    void* p = ws + off;
    off = (off + bytes + 255) & ~(size_t)255;
    return p;
  };
  float*  z0     = (float*)carve((size_t)NS0 * D * 4);
  float*  wz     = (float*)carve((size_t)NS0 * D * 4);
  float*  z1     = (float*)carve((size_t)ND0 * D * 4);
  float*  z2     = (float*)carve((size_t)ND1 * D * 4);
  __bf16* featbf = (__bf16*)carve((size_t)NS0 * F * 2);
  __bf16* z0bf   = (__bf16*)carve((size_t)NS0 * D * 2);
  __bf16* z1bf   = (__bf16*)carve((size_t)NS1p * D * 2);
  __bf16* z2bf   = (__bf16*)carve((size_t)ND1p * D * 2);
  __bf16* WinT   = (__bf16*)carve((size_t)D * F * 2);
  __bf16* W1T    = (__bf16*)carve((size_t)D * D * 2);
  __bf16* W2T    = (__bf16*)carve((size_t)D * D * 2);
  __bf16* WoutT  = (__bf16*)carve((size_t)O * D * 2);
  float*  eeB    = (float*)carve((size_t)E0 * 2 * 4);
  float*  eiB    = (float*)carve((size_t)E0 * 4);
  float*  naB    = (float*)carve((size_t)ND0 * 2 * 4);
  float*  sacc   = (float*)carve((size_t)ND0 * D * 4);
  float*  ehB    = (float*)carve((size_t)ND0 * D * 4);
  float*  ihB    = (float*)carve((size_t)ND0 * D * 4);
  float*  degB   = (float*)carve((size_t)((ND0 + 3) & ~3) * 4);
  (void)ws_size;

  auto waves = [](long nw) { return dim3((unsigned)((nw * 32 + 255) / 256)); };
  auto lin   = [](long n)  { return dim3((unsigned)((n + 255) / 256)); };
  auto lin4  = [](long n)  { return dim3((unsigned)((n / 4 + 255) / 256)); };
  auto gemm_grid = [](int Mtiles) { return dim3(1, (unsigned)((Mtiles + 3) / 4)); };
  auto lds_bytes = [](int K, int N) { return (size_t)N * (K + 8) * 2; };

  // feat -> bf16 once (shared by both modes)
  f32_to_bf16_pad<<<lin4((long)NS0 * F), 256, 0, stream>>>(
      feat, featbf, (long)NS0 * F, (long)NS0 * F);

  for (int m = 0; m < 2; ++m) {
    const int* es0 = (m == 0) ? src0_cor : src0_sim;
    const int* ed0 = (m == 0) ? dst0_cor : dst0_sim;
    const int* is0 = (m == 0) ? src0_sim : src0_cor;
    const int* id0 = (m == 0) ? dst0_sim : dst0_cor;
    const int* es1 = (m == 0) ? src1_cor : src1_sim;
    const int* ed1 = (m == 0) ? dst1_cor : dst1_sim;
    const int* is1 = (m == 0) ? src1_sim : src1_cor;
    const int* id1 = (m == 0) ? dst1_sim : dst1_cor;

    // weight transposes (tiny)
    transpose_to_bf16<<<lin(F * D), 256, 0, stream>>>(Win + (size_t)m * F * D, WinT, F, D);
    transpose_to_bf16<<<lin(D * D), 256, 0, stream>>>(W1  + (size_t)m * D * D, W1T,  D, D);
    transpose_to_bf16<<<lin(D * D), 256, 0, stream>>>(W2  + (size_t)m * D * D, W2T,  D, D);
    transpose_to_bf16<<<lin(D * O), 256, 0, stream>>>(Wout+ (size_t)m * D * O, WoutT,D, O);

    // ---- input projection: z0 = feat @ Win[m] + b_in[m] ----
    gemm_bf16_wmma<8><<<gemm_grid(NS0 / 16), 128, lds_bytes(F, D), stream>>>(
        featbf, WinT, b_in + (size_t)m * D, z0, z0bf, NS0, F, D, NS0 / 16);

    // ---- layer 1 ----
    node_softmax2<<<waves(ND0), 256, 0, stream>>>(z0, na1 + (size_t)m * D * 2, naB, ND0);
    edge_softmax2<<<waves(E0), 256, 0, stream>>>(z0, ea1 + (size_t)m * 2 * D * 2, es0, ed0, eeB, E0);
    edge_sigmoid <<<waves(E0), 256, 0, stream>>>(z0, ia1 + (size_t)m * 2 * D, is0, id0, eiB, E0);
    gemm_bf16_wmma<8><<<gemm_grid(NS0 / 16), 128, lds_bytes(D, D), stream>>>(
        z0bf, W1T, b1 + (size_t)m * D, wz, (__bf16*)nullptr, NS0, D, D, NS0 / 16);

    zero_f32<<<lin4((long)ND0 * D), 256, 0, stream>>>(sacc, ND0 * D);
    zero_f32<<<lin4((long)ND0 * D), 256, 0, stream>>>(ehB,  ND0 * D);
    zero_f32<<<lin4((long)ND0 * D), 256, 0, stream>>>(ihB,  ND0 * D);
    zero_f32<<<lin4((ND0 + 3) & ~3), 256, 0, stream>>>(degB, (ND0 + 3) & ~3);

    scatter_e<<<waves(E0), 256, 0, stream>>>(z0, wz, eeB, es0, ed0, sacc, ehB, degB, E0);
    scatter_i<<<waves(E0), 256, 0, stream>>>(z0, eiB, is0, id0, ihB, E0);
    finalize_layer<<<lin((long)ND0 * 32), 256, 0, stream>>>(z0, naB, sacc, degB, ehB, ihB, z1, ND0);
    f32_to_bf16_pad<<<lin4((long)NS1p * D), 256, 0, stream>>>(
        z1, z1bf, (long)ND0 * D, (long)NS1p * D);

    // ---- layer 2 ----
    node_softmax2<<<waves(ND1), 256, 0, stream>>>(z1, na2 + (size_t)m * D * 2, naB, ND1);
    edge_softmax2<<<waves(E1), 256, 0, stream>>>(z1, ea2 + (size_t)m * 2 * D * 2, es1, ed1, eeB, E1);
    edge_sigmoid <<<waves(E1), 256, 0, stream>>>(z1, ia2 + (size_t)m * 2 * D, is1, id1, eiB, E1);
    gemm_bf16_wmma<8><<<gemm_grid(NS1p / 16), 128, lds_bytes(D, D), stream>>>(
        z1bf, W2T, b2 + (size_t)m * D, wz, (__bf16*)nullptr, NS1, D, D, NS1p / 16);

    zero_f32<<<lin4((long)ND1 * D), 256, 0, stream>>>(sacc, ND1 * D);
    zero_f32<<<lin4((long)ND1 * D), 256, 0, stream>>>(ehB,  ND1 * D);
    zero_f32<<<lin4((long)ND1 * D), 256, 0, stream>>>(ihB,  ND1 * D);
    zero_f32<<<lin4((ND1 + 3) & ~3), 256, 0, stream>>>(degB, (ND1 + 3) & ~3);

    scatter_e<<<waves(E1), 256, 0, stream>>>(z1, wz, eeB, es1, ed1, sacc, ehB, degB, E1);
    scatter_i<<<waves(E1), 256, 0, stream>>>(z1, eiB, is1, id1, ihB, E1);
    finalize_layer<<<lin((long)ND1 * 32), 256, 0, stream>>>(z1, naB, sacc, degB, ehB, ihB, z2, ND1);
    f32_to_bf16_pad<<<lin4((long)ND1p * D), 256, 0, stream>>>(
        z2, z2bf, (long)ND1 * D, (long)ND1p * D);

    // ---- output projection: out[m] = z2 @ Wout[m] + bout[m] ----
    gemm_bf16_wmma<4><<<gemm_grid(ND1p / 16), 128, lds_bytes(D, O), stream>>>(
        z2bf, WoutT, bout + (size_t)m * O, out + (size_t)m * ND1 * O,
        (__bf16*)nullptr, ND1, D, O, ND1p / 16);
  }
}
